// GraphQA_22986664968607
// MI455X (gfx1250) — matrix operand order repaired and checked by reading, not compiled, not run on hardware
//
#include <hip/hip_runtime.h>
#include <hip/hip_bf16.h>

typedef __attribute__((ext_vector_type(16))) _Float16 v16h;
typedef __attribute__((ext_vector_type(8)))  _Float16 v8h;
typedef __attribute__((ext_vector_type(8)))  float    v8f;

#define THREADS 256

__device__ __forceinline__ float seg_load(const void* p, int ish, long long off) {
  return ish ? (float)((const _Float16*)p)[off] : ((const float*)p)[off];
}

// D = act( concat(seg0[i0[r]], seg1[i1[r]], seg2[i2[r]]) @ W + bias )
// Block: 256 thr = 8 waves; 64 rows/block. Wave (rt = wv&3) owns rows
// rt*16..+15; (ng = wv>>2) owns N-tiles ng, ng+2, ...
// lA: 64x32 f16 row-major tile.  lB: W tile pre-swizzled into the exact
// per-lane B-fragment layout so each lane reads one contiguous 32B chunk.
template <int NTS>
__global__ void __launch_bounds__(THREADS) gemm_wmma(
    const void* p0, int w0, const int* i0, int h0,
    const void* p1, int w1, const int* i1, int h1,
    const void* p2, int w2, const int* i2, int h2,
    const float* Wt, const float* bias, int Ktot, int Nout, int act,
    void* out, int ostride, int ohalf, int M)
{
  constexpr int NoutPad = NTS * 16;
  constexpr int JMAX = (NTS + 1) / 2;
  __shared__ __align__(32) _Float16 lA[64 * 32];
  __shared__ __align__(32) _Float16 lB[32 * NoutPad];
  const int tid  = threadIdx.x;
  const int lane = tid & 31;
  const int wv   = tid >> 5;
  const int lm = lane & 15, lh = lane >> 4;
  const int rt = wv & 3, ng = wv >> 2;
  const int mbase = blockIdx.x * 64;
  const int w01 = w0 + w1;
  // A staging assignment: thread -> (row, 8-wide k chunk)
  const int sr = tid >> 2;
  const int skc = (tid & 3) * 8;
  const int srow = mbase + sr;

  v8f acc[JMAX];
#pragma unroll
  for (int j = 0; j < JMAX; ++j)
#pragma unroll
    for (int r = 0; r < 8; ++r) acc[j][r] = 0.f;

  for (int kb = 0; kb < Ktot; kb += 32) {
    // ---- stage A tile (64x32 f16): vector fast path, scalar fallback ----
    {
      const int gk0 = kb + skc;
      v8h av = {};
      if (srow < M) {
        bool done = false;
        if (h0 && gk0 + 8 <= w0) {
          int rr = i0 ? i0[srow] : srow;
          const _Float16* sp = (const _Float16*)p0 + (long long)rr * w0 + gk0;
          if (((size_t)sp & 15) == 0) { av = *(const v8h*)sp; done = true; }
        } else if (h1 && gk0 >= w0 && gk0 + 8 <= w01) {
          int rr = i1 ? i1[srow] : srow;
          const _Float16* sp = (const _Float16*)p1 + (long long)rr * w1 + (gk0 - w0);
          if (((size_t)sp & 15) == 0) { av = *(const v8h*)sp; done = true; }
        } else if (h2 && gk0 >= w01 && gk0 + 8 <= Ktot) {
          int rr = i2 ? i2[srow] : srow;
          const _Float16* sp = (const _Float16*)p2 + (long long)rr * w2 + (gk0 - w01);
          if (((size_t)sp & 15) == 0) { av = *(const v8h*)sp; done = true; }
        }
        if (!done) {
#pragma unroll
          for (int e = 0; e < 8; ++e) {
            int gk = gk0 + e;
            float v = 0.f;
            if (gk < w0)        { int rr = i0 ? i0[srow] : srow; v = seg_load(p0, h0, (long long)rr * w0 + gk); }
            else if (gk < w01)  { int rr = i1 ? i1[srow] : srow; v = seg_load(p1, h1, (long long)rr * w1 + (gk - w0)); }
            else if (gk < Ktot) { int rr = i2 ? i2[srow] : srow; v = seg_load(p2, h2, (long long)rr * w2 + (gk - w01)); }
            av[e] = (_Float16)v;
          }
        }
      }
      *(v8h*)&lA[sr * 32 + skc] = av;
    }
    // ---- stage W tile, swizzled to B-fragment layout [nt][lane][e] ----
    for (int idx = tid; idx < 32 * NoutPad; idx += THREADS) {
      int k = idx / NoutPad, n = idx - k * NoutPad;
      int gk = kb + k;
      float v = (gk < Ktot && n < Nout) ? Wt[(long long)gk * Nout + n] : 0.f;
      int dlane = ((k & 8) << 1) + (n & 15);         // lanes 0-15: K{0..7,16..23}, 16-31: K{8..15,24..31}
      lB[(((n >> 4) * 32) + dlane) * 16 + ((k & 7) | ((k & 16) >> 1))] = (_Float16)v;
    }
    __syncthreads();

    // A fragment: lanes 0-15 K{0..7,16..23}, lanes 16-31 K{8..15,24..31}
    const _Float16* ap = &lA[(rt * 16 + lm) * 32 + (lh ? 8 : 0)];
    v8h alo = *(const v8h*)ap;
    v8h ahi = *(const v8h*)(ap + 16);
    v16h a = __builtin_shufflevector(alo, ahi,
        0, 1, 2, 3, 4, 5, 6, 7, 8, 9, 10, 11, 12, 13, 14, 15);

#pragma unroll
    for (int jj = 0; jj < JMAX; ++jj) {
      int nt = jj * 2 + ng;
      if ((NTS & 1) == 0 || nt < NTS) {        // compile-time true for even NTS
        v16h b = *(const v16h*)&lB[(nt * 32 + lane) * 16];
        acc[jj] = __builtin_amdgcn_wmma_f32_16x16x32_f16(
            false, a, false, b, (short)0, acc[jj], false, false);
      }
    }
    __syncthreads();
  }

  // ---- epilogue: bias + activation + store (f16 or f32, strided) ----
#pragma unroll
  for (int jj = 0; jj < JMAX; ++jj) {
    int nt = jj * 2 + ng;
    if ((NTS & 1) == 0 || nt < NTS) {
      int col = nt * 16 + lm;
#pragma unroll
      for (int r = 0; r < 8; ++r) {
        int row = mbase + rt * 16 + r + (lh ? 8 : 0);
        if (row < M && col < Nout) {
          float v = acc[jj][r] + bias[col];
          if (act == 1)      v = fmaxf(v, 0.f);
          else if (act == 2) v = 1.f / (1.f + __expf(-v));
          long long o = (long long)row * ostride + col;
          if (ohalf) ((_Float16*)out)[o] = (_Float16)v;
          else       ((float*)out)[o] = v;
        }
      }
    }
  }
}

__global__ void zerof(float* p, long long n) {
  long long t = (long long)blockIdx.x * blockDim.x + threadIdx.x;
  if (t < n) p[t] = 0.f;
}

__global__ void scatter_add(const void* vals, int vh, const int* idx,
                            float* out, float* cnt, long long R, int F) {
  long long t = (long long)blockIdx.x * blockDim.x + threadIdx.x;
  if (t >= R * (long long)F) return;
  long long r = t / F;
  int f = (int)(t - r * F);
  float v = vh ? (float)((const _Float16*)vals)[t] : ((const float*)vals)[t];
  int s = idx[r];
  atomicAdd(out + (long long)s * F + f, v);
  if (f == 0) atomicAdd(cnt + s, 1.f);
}

__global__ void div_mean(float* out, const float* cnt, long long S, int F) {
  long long t = (long long)blockIdx.x * blockDim.x + threadIdx.x;
  if (t >= S * (long long)F) return;
  out[t] = out[t] / fmaxf(cnt[t / F], 1.f);
}

__global__ void assemble_xh(_Float16* X0, const int* aa, const int* ss,
                            const float* aae, const float* sse, int N) {
  long long t = (long long)blockIdx.x * blockDim.x + threadIdx.x;
  if (t >= (long long)N * 128) return;
  int n = (int)(t >> 7);
  int j = (int)(t & 127);
  float v = (j < 64) ? aae[aa[n] * 64 + j] : sse[ss[n] * 64 + (j - 64)];
  X0[(long long)n * 256 + 128 + j] = (_Float16)v;
}

__global__ void assemble_eh(_Float16* eh, const int* src, const int* dst,
                            const float* sepe, const float* dist, const float* rbfc, int E) {
  long long t = (long long)blockIdx.x * blockDim.x + threadIdx.x;
  if (t >= (long long)E * 80) return;
  int e = (int)(t / 80);
  int j = (int)(t - (long long)e * 80);
  float v; int col;
  if (j < 64) {
    int d = src[e] - dst[e]; if (d < 0) d = -d;
    int code = (d > 1) + (d > 2) + (d > 3) + (d > 4) + (d > 5) + (d > 10) + (d > 15);
    v = sepe[code * 64 + j]; col = 64 + j;
  } else {
    float c = rbfc[j - 64]; float dd = dist[e] - c;
    v = __expf(-dd * dd); col = 128 + (j - 64);
  }
  eh[(long long)e * 144 + col] = (_Float16)v;
}

__global__ void make_bsrc(int* bsrc, const int* src, const int* batch, int E) {
  int t = blockIdx.x * blockDim.x + threadIdx.x;
  if (t < E) bsrc[t] = batch[src[t]];
}

extern "C" void kernel_launch(void* const* d_in, const int* in_sizes, int n_in,
                              void* d_out, int out_size, void* d_ws, size_t ws_size,
                              hipStream_t stream)
{
  (void)n_in; (void)out_size; (void)ws_size;
  const int N = in_sizes[0] / 3;
  const int E = in_sizes[3];
  const int G = 32;

  const float* x     = (const float*)d_in[0];
  const float* msa   = (const float*)d_in[1];
  const float* eattr = (const float*)d_in[2];
  const float* dist  = (const float*)d_in[3];
  const int*   aa    = (const int*)d_in[4];
  const int*   ss    = (const int*)d_in[5];
  const int*   eidx  = (const int*)d_in[6];
  const int*   batch = (const int*)d_in[7];
  const float* aae   = (const float*)d_in[8];
  const float* sse   = (const float*)d_in[9];
  const float* sepe  = (const float*)d_in[10];
  const float* rbfc  = (const float*)d_in[11];
  const int* src = eidx;
  const int* dst = eidx + E;

  char* ws = (char*)d_ws;
  size_t off = 0;
  auto alloc = [&](size_t bytes) -> void* {
    void* p = ws + off;
    off += (bytes + 255) & ~(size_t)255;
    return p;
  };
  _Float16* X0  = (_Float16*)alloc((size_t)N * 256 * 2);
  _Float16* xhA = (_Float16*)alloc((size_t)N * 256 * 2);
  _Float16* xhB = (_Float16*)alloc((size_t)N * 256 * 2);
  _Float16* ehA = (_Float16*)alloc((size_t)E * 144 * 2);
  _Float16* ehB = (_Float16*)alloc((size_t)E * 128 * 2);
  _Float16* uA  = (_Float16*)alloc((size_t)G * 256 * 2);
  _Float16* uB  = (_Float16*)alloc((size_t)G * 256 * 2);
  float* msg  = (float*)alloc((size_t)N * 128 * 4);
  float* eg   = (float*)alloc((size_t)G * 128 * 4);
  float* xg   = (float*)alloc((size_t)G * 256 * 4);
  float* cntN = (float*)alloc((size_t)N * 4);
  float* cntG = (float*)alloc((size_t)G * 4);
  int*   bsrc = (int*)alloc((size_t)E * 4);
  _Float16* nh1 = xhB;  // alias: dead before xhB first written (layer 2)
  _Float16* eh1 = ehB;  // alias: dead before ehB first written (layer 1)

  auto gemm = [&](const void* p0, int w0, const int* i0, int h0,
                  const void* p1, int w1, const int* i1, int h1,
                  const void* p2, int w2, const int* i2, int h2,
                  const float* Wt, const float* b, int Nout, int act,
                  void* outp, int ostr, int ohalf, int M) {
    int nts = (Nout + 15) >> 4;
    dim3 g((M + 63) / 64), blk(THREADS);
    int Kt = w0 + w1 + w2;
    if (nts <= 1)
      gemm_wmma<1><<<g, blk, 0, stream>>>(p0,w0,i0,h0,p1,w1,i1,h1,p2,w2,i2,h2,Wt,b,Kt,Nout,act,outp,ostr,ohalf,M);
    else if (nts <= 2)
      gemm_wmma<2><<<g, blk, 0, stream>>>(p0,w0,i0,h0,p1,w1,i1,h1,p2,w2,i2,h2,Wt,b,Kt,Nout,act,outp,ostr,ohalf,M);
    else if (nts <= 4)
      gemm_wmma<4><<<g, blk, 0, stream>>>(p0,w0,i0,h0,p1,w1,i1,h1,p2,w2,i2,h2,Wt,b,Kt,Nout,act,outp,ostr,ohalf,M);
    else if (nts <= 8)
      gemm_wmma<8><<<g, blk, 0, stream>>>(p0,w0,i0,h0,p1,w1,i1,h1,p2,w2,i2,h2,Wt,b,Kt,Nout,act,outp,ostr,ohalf,M);
    else
      gemm_wmma<16><<<g, blk, 0, stream>>>(p0,w0,i0,h0,p1,w1,i1,h1,p2,w2,i2,h2,Wt,b,Kt,Nout,act,outp,ostr,ohalf,M);
  };
  auto zero = [&](float* p, long long n) {
    zerof<<<dim3((unsigned)((n + 255) / 256)), dim3(256), 0, stream>>>(p, n);
  };
  auto smean = [&](const void* vals, int vh, const int* idx, float* o, float* c,
                   long long R, int F, long long S) {
    zero(o, S * F);
    zero(c, S);
    long long tot = R * F;
    scatter_add<<<dim3((unsigned)((tot + 255) / 256)), dim3(256), 0, stream>>>(vals, vh, idx, o, c, R, F);
    long long tot2 = S * F;
    div_mean<<<dim3((unsigned)((tot2 + 255) / 256)), dim3(256), 0, stream>>>(o, c, S, F);
  };

  // ---------------- Encoders ----------------
  gemm(x, 3, nullptr, 0, msa, 21, nullptr, 0, nullptr, 0, nullptr, 0,
       (const float*)d_in[12], (const float*)d_in[13], 64, 1, nh1, 64, 1, N);
  gemm(nh1, 64, nullptr, 1, nullptr, 0, nullptr, 0, nullptr, 0, nullptr, 0,
       (const float*)d_in[14], (const float*)d_in[15], 128, 1, X0, 256, 1, N);
  {
    long long tot = (long long)N * 128;
    assemble_xh<<<dim3((unsigned)((tot + 255) / 256)), dim3(256), 0, stream>>>(X0, aa, ss, aae, sse, N);
  }
  gemm(eattr, 4, nullptr, 0, nullptr, 0, nullptr, 0, nullptr, 0, nullptr, 0,
       (const float*)d_in[16], (const float*)d_in[17], 32, 1, eh1, 32, 1, E);
  gemm(eh1, 32, nullptr, 1, nullptr, 0, nullptr, 0, nullptr, 0, nullptr, 0,
       (const float*)d_in[18], (const float*)d_in[19], 64, 1, ehA, 144, 1, E);
  {
    long long tot = (long long)E * 80;
    assemble_eh<<<dim3((unsigned)((tot + 255) / 256)), dim3(256), 0, stream>>>(ehA, src, dst, sepe, dist, rbfc, E);
  }
  make_bsrc<<<dim3((E + 255) / 256), dim3(256), 0, stream>>>(bsrc, src, batch, E);

  // ---------------- Message-passing layers ----------------
  static const int LS[6][6] = {
      {144, 256, 0, 128, 256, 256}, {128, 256, 256, 64, 128, 256},
      {64, 128, 256, 64, 128, 128}, {64, 128, 128, 32, 128, 64},
      {32, 128, 64, 32, 64, 64},    {32, 64, 64, 16, 64, 32}};

  const _Float16* xin = X0;  _Float16* xout = xhA;
  const _Float16* ein = ehA; _Float16* eout = ehB;
  const _Float16* uin = nullptr; _Float16* uout = uA;
  for (int k = 0; k < 6; ++k) {
    const int ie = LS[k][0], inn = LS[k][1], ig = LS[k][2];
    const int oe = LS[k][3], on = LS[k][4], og = LS[k][5];
    const float* We = (const float*)d_in[20 + 6 * k + 0];
    const float* be = (const float*)d_in[20 + 6 * k + 1];
    const float* Wn = (const float*)d_in[20 + 6 * k + 2];
    const float* bn = (const float*)d_in[20 + 6 * k + 3];
    const float* Wg = (const float*)d_in[20 + 6 * k + 4];
    const float* bg = (const float*)d_in[20 + 6 * k + 5];

    // edge update: relu(concat(xh[src], eh, u[batch_src]) @ We + be)
    gemm(xin, inn, src, 1, ein, ie, nullptr, 1, uin, ig, bsrc, 1,
         We, be, oe, 1, eout, oe, 1, E);
    // node update: msg = scatter_mean(eh', dst); relu(concat(xh, msg, u[batch]) @ Wn + bn)
    smean(eout, 1, dst, msg, cntN, E, oe, N);
    gemm(xin, inn, nullptr, 1, msg, oe, nullptr, 0, uin, ig, batch, 1,
         Wn, bn, on, 1, xout, on, 1, N);
    // global update
    smean(eout, 1, bsrc, eg, cntG, E, oe, G);
    smean(xout, 1, batch, xg, cntG, N, on, G);
    gemm(eg, oe, nullptr, 0, xg, on, nullptr, 0, uin, ig, nullptr, 1,
         Wg, bg, og, 1, uout, og, 1, G);

    ein = eout; eout = (eout == ehA) ? ehB : ehA;
    xin = xout; xout = (xout == xhA) ? xhB : xhA;
    uin = uout; uout = (uout == uA) ? uB : uA;
  }

  // ---------------- Readouts ----------------
  gemm(xin, 64, nullptr, 1, X0, 256, nullptr, 1, nullptr, 0, nullptr, 0,
       (const float*)d_in[56], (const float*)d_in[57], 2, 2, d_out, 2, 0, N);
  gemm(uin, 32, nullptr, 1, nullptr, 0, nullptr, 0, nullptr, 0, nullptr, 0,
       (const float*)d_in[58], (const float*)d_in[59], 5, 2,
       (float*)d_out + (size_t)N * 2, 5, 0, G);
}
